// MOEBlock_11879879544252
// MI455X (gfx1250) — matrix operand order repaired and checked by reading, not compiled, not run on hardware
//
#include <hip/hip_runtime.h>
#include <hip/hip_bf16.h>
#include <math.h>

typedef __attribute__((ext_vector_type(16))) __bf16 v16bf;
typedef __attribute__((ext_vector_type(8)))  __bf16 v8bf;
typedef __attribute__((ext_vector_type(8)))  float  v8f;

static constexpr int N_   = 8;
static constexpr int C_   = 384;
static constexpr int H_   = 56;
static constexpr int W_   = 56;
static constexpr int C4_  = 96;
static constexpr int E_   = 8;
static constexpr int R_   = 6;
static constexpr int HW_  = H_ * W_;       // 3136
static constexpr int P_   = N_ * HW_;      // 25088
static constexpr float EPSF = 1e-5f;
static constexpr int OUT_MAIN = N_ * C_ * HW_;   // 9,633,792 floats
static constexpr int OUT_RF   = N_ * (2 * C4_);  // 1,536 floats (route_feat)

// ---------------- device helpers ----------------

__device__ __forceinline__ float geluf(float x) {
  return 0.5f * x * (1.0f + erff(x * 0.7071067811865475f));
}

__device__ __forceinline__ float wave_sum(float v) {
#pragma unroll
  for (int m = 16; m >= 1; m >>= 1) v += __shfl_xor(v, m, 32);
  return v;
}

// A fragment (16x32 bf16): lane = (g, m); element e -> K = g*8 + e (+8 for e>=8)
__device__ __forceinline__ v16bf load_a_frag(const __bf16* __restrict__ row_k0, int g) {
  v8bf lo = *(const v8bf*)(row_k0 + g * 8);
  v8bf hi = *(const v8bf*)(row_k0 + 16 + g * 8);
  v16bf a;
#pragma unroll
  for (int i = 0; i < 8; ++i) { a[i] = lo[i]; a[i + 8] = hi[i]; }
  return a;
}

// B fragment (32x16 bf16): lane = (g, n); element e -> K = g*8 + e (+8 for e>=8)
// wrow_k0 points at row (o0+n) of W[Cout x K], at column k0.
__device__ __forceinline__ v16bf load_b_frag(const __bf16* __restrict__ wrow_k0, int g) {
  v8bf lo = *(const v8bf*)(wrow_k0 + g * 8);
  v8bf hi = *(const v8bf*)(wrow_k0 + 16 + g * 8);
  v16bf b;
#pragma unroll
  for (int i = 0; i < 8; ++i) { b[i] = lo[i]; b[i + 8] = hi[i]; }
  return b;
}

// ---------------- generic bf16 WMMA GEMM, NG col-tiles per wave ----------------
// Out[p, o] = sum_k A[p, k] * W[o, k] + bias[o] + biasNC[n, o]
// Each wave computes a 16 x (NG*16) strip: A fragment loaded once per k-step,
// reused against NG B fragments (2.5 loads/WMMA at NG=4 vs 4 without blocking).
template <int NG>
__global__ void k_gemm(const __bf16* __restrict__ A, int lda,
                       const __bf16* __restrict__ Wm, int ldw, int wStride,
                       const float* __restrict__ bias,
                       const float* __restrict__ biasNC,
                       float* __restrict__ Of, __bf16* __restrict__ Ob,
                       int ldo, int Prows, int Cout, int K, int outNCHW) {
  int wave = blockIdx.x * (blockDim.x >> 5) + (threadIdx.x >> 5);
  int colG = (Cout >> 4) / NG;          // column groups
  int rowT = Prows >> 4;
  if (wave >= rowT * colG) return;
  int p0 = (wave / colG) << 4;
  int o0 = (wave % colG) * (NG * 16);
  int lane = threadIdx.x & 31, g = lane >> 4, mn = lane & 15;
  int nb = p0 / HW_;   // batch index (tiles never cross n: HW_ % 16 == 0)

  const __bf16* arow = A + (size_t)(p0 + mn) * lda;
  const __bf16* wrow[NG];
#pragma unroll
  for (int q = 0; q < NG; ++q)
    wrow[q] = Wm + (size_t)nb * wStride + (size_t)(o0 + q * 16 + mn) * ldw;

  v8f acc[NG];
#pragma unroll
  for (int q = 0; q < NG; ++q) acc[q] = {};

  for (int k0 = 0; k0 < K; k0 += 32) {
    v16bf a = load_a_frag(arow + k0, g);
#pragma unroll
    for (int q = 0; q < NG; ++q) {
      v16bf b = load_b_frag(wrow[q] + k0, g);
      acc[q] = __builtin_amdgcn_wmma_f32_16x16x32_bf16(false, a, false, b, (short)0,
                                                       acc[q], false, false);
    }
  }

#pragma unroll
  for (int q = 0; q < NG; ++q) {
    int oc = o0 + q * 16 + mn;
    float bc  = bias   ? bias[oc] : 0.0f;
    float bnc = biasNC ? biasNC[(size_t)nb * Cout + oc] : 0.0f;
#pragma unroll
    for (int r = 0; r < 8; ++r) {
      int row = p0 + g * 8 + r;
      float v = acc[q][r] + bc + bnc;
      if (Of) {
        if (outNCHW) {
          int n = row / HW_, hw = row % HW_;
          Of[((size_t)n * Cout + oc) * HW_ + hw] = v;
        } else {
          Of[(size_t)row * ldo + oc] = v;
        }
      }
      if (Ob) Ob[(size_t)row * ldo + oc] = (__bf16)v;
    }
  }
}

// ---------------- 3x3 SAME conv as 9 shifted WMMA GEMMs ----------------
// A: out_ori bf16 [P, C]; W9: bf16 [9][C][C] tap-major; zrow: C_ bf16 zeros.
// Each wave computes a 16 x 64 strip (4 col-tiles). Halo handling is branchless:
// out-of-image lanes read from the all-zeros row via a pointer select, so the
// load stream is uniform and can software-pipeline across taps (no EXEC games).
__global__ void k_conv3x3(const __bf16* __restrict__ A, const __bf16* __restrict__ W9,
                          const float* __restrict__ bias,
                          const __bf16* __restrict__ zrow, float* __restrict__ Out) {
  constexpr int NG = 4;
  int wave = blockIdx.x * (blockDim.x >> 5) + (threadIdx.x >> 5);
  const int colG = (C_ / 16) / NG, rowT = P_ / 16;
  if (wave >= rowT * colG) return;
  int p0 = (wave / colG) << 4;
  int o0 = (wave % colG) * (NG * 16);
  int lane = threadIdx.x & 31, g = lane >> 4, mn = lane & 15;
  int pl = p0 + mn;
  int hw = pl % HW_;
  int hh = hw / W_, ww = hw % W_;

  // Per-tap source rows, resolved branchlessly up front.
  const __bf16* arows[9];
#pragma unroll
  for (int t = 0; t < 9; ++t) {
    const int dy = t / 3 - 1, dx = t % 3 - 1;
    bool valid = ((unsigned)(hh + dy) < (unsigned)H_) &&
                 ((unsigned)(ww + dx) < (unsigned)W_);
    arows[t] = valid ? (A + (size_t)(pl + dy * W_ + dx) * C_) : zrow;
  }

  v8f acc[NG];
#pragma unroll
  for (int q = 0; q < NG; ++q) acc[q] = {};

#pragma unroll
  for (int t = 0; t < 9; ++t) {
    const __bf16* arow = arows[t];
    const __bf16* wbase = W9 + (size_t)t * C_ * C_ + (size_t)(o0 + mn) * C_;
    for (int k0 = 0; k0 < C_; k0 += 32) {
      v16bf a = load_a_frag(arow + k0, g);
#pragma unroll
      for (int q = 0; q < NG; ++q) {
        v16bf b = load_b_frag(wbase + (size_t)(q * 16) * C_ + k0, g);
        acc[q] = __builtin_amdgcn_wmma_f32_16x16x32_bf16(false, a, false, b, (short)0,
                                                         acc[q], false, false);
      }
    }
  }
#pragma unroll
  for (int q = 0; q < NG; ++q) {
    int oc = o0 + q * 16 + mn;
    float bc = bias[oc];
#pragma unroll
    for (int r = 0; r < 8; ++r) {
      int row = p0 + g * 8 + r;
      Out[(size_t)row * C_ + oc] = acc[q][r] + bc;
    }
  }
}

// ---------------- layout / conversion kernels ----------------

// NCHW fp32 -> pixel-major [P, C] bf16
__global__ void k_cvt_x(const float* __restrict__ x, __bf16* __restrict__ xt) {
  int idx = blockIdx.x * blockDim.x + threadIdx.x;
  if (idx >= N_ * C_ * HW_) return;
  int n = idx / (C_ * HW_);
  int rem = idx % (C_ * HW_);
  int c = rem / HW_;
  int hw = rem % HW_;
  xt[(size_t)(n * HW_ + hw) * C_ + c] = (__bf16)x[idx];
}

// fp32 [srows, scols] -> bf16 [drows, dcols], zero-padded
__global__ void k_padcvt(const float* __restrict__ src, int srows, int scols,
                         __bf16* __restrict__ dst, int drows, int dcols) {
  int idx = blockIdx.x * blockDim.x + threadIdx.x;
  if (idx >= drows * dcols) return;
  int r = idx / dcols, c = idx % dcols;
  float v = (r < srows && c < scols) ? src[(size_t)r * scols + c] : 0.0f;
  dst[idx] = (__bf16)v;
}

// share_w (O,I,3,3) fp32 -> [t][O][I] bf16
__global__ void k_repack_shr(const float* __restrict__ w, __bf16* __restrict__ dst) {
  int idx = blockIdx.x * blockDim.x + threadIdx.x;
  if (idx >= 9 * C_ * C_) return;
  int t = idx / (C_ * C_);
  int rem = idx % (C_ * C_);
  int o = rem / C_, i = rem % C_;
  dst[idx] = (__bf16)w[((size_t)o * C_ + i) * 9 + t];
}

__global__ void k_padf(const float* __restrict__ src, int nsrc,
                       float* __restrict__ dst, int ndst) {
  int i = blockIdx.x * blockDim.x + threadIdx.x;
  if (i < ndst) dst[i] = (i < nsrc) ? src[i] : 0.0f;
}

// ---------------- pointwise / reduction stages (fp32) ----------------

// per-pixel LN over 96 channels + exact GELU -> bf16
__global__ void k_ln96_gelu(const float* __restrict__ feat,
                            const float* __restrict__ g, const float* __restrict__ b,
                            __bf16* __restrict__ out) {
  int wid = threadIdx.x >> 5, lane = threadIdx.x & 31;
  int p = blockIdx.x * (blockDim.x >> 5) + wid;
  if (p >= P_) return;
  const float* row = feat + (size_t)p * C4_;
  float x0 = row[lane], x1 = row[lane + 32], x2 = row[lane + 64];
  float mean = wave_sum(x0 + x1 + x2) * (1.0f / 96.0f);
  float d0 = x0 - mean, d1 = x1 - mean, d2 = x2 - mean;
  float var = wave_sum(d0 * d0 + d1 * d1 + d2 * d2) * (1.0f / 96.0f);
  float rstd = rsqrtf(var + EPSF);
  __bf16* orow = out + (size_t)p * C4_;
  orow[lane]      = (__bf16)geluf(d0 * rstd * g[lane]      + b[lane]);
  orow[lane + 32] = (__bf16)geluf(d1 * rstd * g[lane + 32] + b[lane + 32]);
  orow[lane + 64] = (__bf16)geluf(d2 * rstd * g[lane + 64] + b[lane + 64]);
}

// fs[n,c] = sum_hw out_ori[n,c,hw] * sp_w[hw] + sp_b
__global__ void k_fs(const __bf16* __restrict__ oo, const float* __restrict__ spw,
                     const float* __restrict__ spb, float* __restrict__ fs) {
  int n = blockIdx.x, c = threadIdx.x;   // blockDim = 384
  const __bf16* base = oo + (size_t)n * HW_ * C_ + c;
  float s = 0.0f;
  for (int hw = 0; hw < HW_; ++hw) s += (float)base[(size_t)hw * C_] * spw[hw];
  fs[n * C_ + c] = s + spb[0];
}

// LN over 384 channels per n + GELU (fp32)
__global__ void k_fs_ln(const float* __restrict__ fs, const float* __restrict__ g,
                        const float* __restrict__ b, float* __restrict__ out) {
  __shared__ float red[12];
  __shared__ float stats[2];
  int n = blockIdx.x, t = threadIdx.x;   // blockDim = 384
  int wid = t >> 5, lane = t & 31;
  float x = fs[n * C_ + t];
  float s = wave_sum(x);
  if (lane == 0) red[wid] = s;
  __syncthreads();
  if (t == 0) {
    float tot = 0.0f;
    for (int i = 0; i < 12; ++i) tot += red[i];
    stats[0] = tot * (1.0f / (float)C_);
  }
  __syncthreads();
  float d = x - stats[0];
  float vs = wave_sum(d * d);
  __syncthreads();
  if (lane == 0) red[wid] = vs;
  __syncthreads();
  if (t == 0) {
    float tot = 0.0f;
    for (int i = 0; i < 12; ++i) tot += red[i];
    stats[1] = rsqrtf(tot * (1.0f / (float)C_) + EPSF);
  }
  __syncthreads();
  out[n * C_ + t] = geluf(d * stats[1] * g[t] + b[t]);
}

// fs2 = fsg @ r_conv2_w^T + b   (tiny, fp32)
__global__ void k_fs2(const float* __restrict__ fsg, const float* __restrict__ w,
                      const float* __restrict__ b, float* __restrict__ fs2) {
  int n = blockIdx.x, o = threadIdx.x;   // blockDim = 96
  float s = b[o];
  for (int j = 0; j < C_; ++j) s += w[(size_t)o * C_ + j] * fsg[n * C_ + j];
  fs2[n * C4_ + o] = s;
}

// mean over spatial of featc [P,96] -> [N,96]
__global__ void k_meanc(const float* __restrict__ featc, float* __restrict__ meanc) {
  int n = blockIdx.x, c = threadIdx.x;   // blockDim = 96
  float s = 0.0f;
  const float* base = featc + (size_t)n * HW_ * C4_ + c;
  for (int hw = 0; hw < HW_; ++hw) s += base[(size_t)hw * C4_];
  meanc[n * C4_ + c] = s * (1.0f / (float)HW_);
}

// router + top-k mask + softmax; also writes route_feat & route to d_out
__global__ void k_router(const float* __restrict__ meanc, const float* __restrict__ fs2,
                         const float* __restrict__ rw, const float* __restrict__ rb,
                         float* __restrict__ route, float* __restrict__ pmix,
                         float* __restrict__ outp) {
  __shared__ float rf[2 * C4_];
  __shared__ float pr[2 * E_ + 1];
  int n = blockIdx.x, lane = threadIdx.x;   // blockDim = 32
  for (int j = lane; j < 2 * C4_; j += 32) {
    float v = (j < C4_) ? meanc[n * C4_ + j] : fs2[n * C4_ + (j - C4_)];
    rf[j] = v;
    outp[OUT_MAIN + n * (2 * C4_) + j] = v;
  }
  __syncthreads();
  for (int o = lane; o < 2 * E_ + 1; o += 32) {
    float s = rb[o];
    for (int j = 0; j < 2 * C4_; ++j) s += rw[(size_t)o * (2 * C4_) + j] * rf[j];
    pr[o] = s;
  }
  __syncthreads();
  if (lane == 0) {
    float raw[E_];
    for (int e = 0; e < E_; ++e) raw[e] = pr[e];
    for (int k = 0; k < E_ - 3; ++k) {   // mask the 5 lowest
      int mi = 0; float mv = 1e30f;
      for (int e = 0; e < E_; ++e)
        if (raw[e] > -1e9f && raw[e] < mv) { mv = raw[e]; mi = e; }
      raw[mi] = -1e10f;
    }
    float mx = -1e30f;
    for (int e = 0; e < E_; ++e) mx = fmaxf(mx, raw[e]);
    float ex[E_], se = 0.0f;
    for (int e = 0; e < E_; ++e) { ex[e] = expf(raw[e] - mx); se += ex[e]; }
    for (int e = 0; e < E_; ++e) {
      float r = ex[e] / se;
      route[n * E_ + e] = r;
      outp[OUT_MAIN + OUT_RF + n * E_ + e] = r;
    }
    pmix[n] = pr[2 * E_];
  }
}

// per-n combined LoRA weight: Wn[n][c][j] = lora_M[e,c,r]*route[n,e], j=e*R+r (<48), else 0
__global__ void k_wn(const float* __restrict__ loraM, const float* __restrict__ route,
                     __bf16* __restrict__ wn) {
  int idx = blockIdx.x * blockDim.x + threadIdx.x;
  if (idx >= N_ * C_ * 64) return;
  int n = idx / (C_ * 64);
  int rem = idx % (C_ * 64);
  int c = rem / 64, j = rem % 64;
  float v = 0.0f;
  if (j < E_ * R_) {
    int e = j / R_, r = j % R_;
    v = loraM[((size_t)e * C_ + c) * R_ + r] * route[n * E_ + e];
  }
  wn[idx] = (__bf16)v;
}

__global__ void k_biasnc(const float* __restrict__ lora_bM, const float* __restrict__ route,
                         float* __restrict__ bnc) {
  int idx = blockIdx.x * blockDim.x + threadIdx.x;
  if (idx >= N_ * C_) return;
  int n = idx / C_, c = idx % C_;
  float s = 0.0f;
  for (int e = 0; e < E_; ++e) s += lora_bM[(size_t)e * C_ + c] * route[n * E_ + e];
  bnc[idx] = s;
}

// BN_all(lora) + spe*prob_mix + share -> BN -> GELU -> bf16
__global__ void k_combine(const float* __restrict__ lora, const float* __restrict__ spe,
                          const float* __restrict__ shr, const float* __restrict__ pmix,
                          const float* __restrict__ bag, const float* __restrict__ bab,
                          const float* __restrict__ bam, const float* __restrict__ bav,
                          const float* __restrict__ bg, const float* __restrict__ bb,
                          const float* __restrict__ bm, const float* __restrict__ bv,
                          __bf16* __restrict__ y) {
  int idx = blockIdx.x * blockDim.x + threadIdx.x;
  if (idx >= P_ * C_) return;
  int p = idx / C_, c = idx % C_;
  int n = p / HW_;
  float a = (lora[idx] - bam[c]) * rsqrtf(bav[c] + EPSF) * bag[c] + bab[c];
  float o = a + spe[idx] * pmix[n] + shr[idx];
  float o2 = (o - bm[c]) * rsqrtf(bv[c] + EPSF) * bg[c] + bb[c];
  y[idx] = (__bf16)geluf(o2);
}

// ---------------- host launcher ----------------

extern "C" void kernel_launch(void* const* d_in, const int* in_sizes, int n_in,
                              void* d_out, int out_size, void* d_ws, size_t ws_size,
                              hipStream_t stream) {
  (void)in_sizes; (void)n_in; (void)out_size; (void)ws_size;
  const float* x        = (const float*)d_in[0];
  const float* conv1_w  = (const float*)d_in[1];
  const float* conv1_b  = (const float*)d_in[2];
  const float* rc1_w    = (const float*)d_in[3];
  const float* rc1_b    = (const float*)d_in[4];
  const float* rln_g    = (const float*)d_in[5];
  const float* rln_b    = (const float*)d_in[6];
  const float* rsp_w    = (const float*)d_in[7];
  const float* rsp_b    = (const float*)d_in[8];
  const float* rlnsp_g  = (const float*)d_in[9];
  const float* rlnsp_b  = (const float*)d_in[10];
  const float* rc2_w    = (const float*)d_in[11];
  const float* rc2_b    = (const float*)d_in[12];
  const float* rc3_w    = (const float*)d_in[13];
  const float* rc3_b    = (const float*)d_in[14];
  const float* router_w = (const float*)d_in[15];
  const float* router_b = (const float*)d_in[16];
  const float* lora_W   = (const float*)d_in[17];
  const float* lora_bW  = (const float*)d_in[18];
  const float* lora_M   = (const float*)d_in[19];
  const float* lora_bM  = (const float*)d_in[20];
  const float* bn_all_g = (const float*)d_in[21];
  const float* bn_all_b = (const float*)d_in[22];
  const float* bn_all_m = (const float*)d_in[23];
  const float* bn_all_v = (const float*)d_in[24];
  const float* spe_W    = (const float*)d_in[25];
  const float* spe_bW   = (const float*)d_in[26];
  const float* spe_M    = (const float*)d_in[27];
  const float* spe_bM   = (const float*)d_in[28];
  const float* share_w  = (const float*)d_in[29];
  const float* share_b  = (const float*)d_in[30];
  const float* bn_g     = (const float*)d_in[31];
  const float* bn_b     = (const float*)d_in[32];
  const float* bn_m     = (const float*)d_in[33];
  const float* bn_v     = (const float*)d_in[34];
  const float* conv3_w  = (const float*)d_in[35];
  const float* conv3_b  = (const float*)d_in[36];

  char* ws = (char*)d_ws;
  size_t off = 0;
  auto alloc = [&](size_t bytes) -> char* {
    char* p = ws + off;
    off = (off + bytes + 255) & ~(size_t)255;
    return p;
  };

  __bf16* XT    = (__bf16*)alloc((size_t)P_ * C_ * 2);   // later reused as Ybf
  __bf16* OObf  = (__bf16*)alloc((size_t)P_ * C_ * 2);
  float*  BIGD  = (float*) alloc((size_t)P_ * C_ * 4);   // feat [P,96] then lora_sum [P,C]
  float*  SPE   = (float*) alloc((size_t)P_ * C_ * 4);
  float*  SHR   = (float*) alloc((size_t)P_ * C_ * 4);
  __bf16* FEATG = (__bf16*)alloc((size_t)P_ * C4_ * 2);  // later reused as z_spe [P,32]
  float*  FEATC = (float*) alloc((size_t)P_ * C4_ * 4);  // later reused as z bf16 [P,64]
  __bf16* Wc1   = (__bf16*)alloc((size_t)C_ * C_ * 2);
  __bf16* Wrc1  = (__bf16*)alloc((size_t)C4_ * C_ * 2);
  __bf16* Wrc3  = (__bf16*)alloc((size_t)C4_ * C4_ * 2);
  __bf16* Wlora = (__bf16*)alloc((size_t)64 * C_ * 2);
  __bf16* Wspe1 = (__bf16*)alloc((size_t)32 * C_ * 2);
  __bf16* Wspe2 = (__bf16*)alloc((size_t)C_ * 32 * 2);
  __bf16* Wshr  = (__bf16*)alloc((size_t)9 * C_ * C_ * 2);
  __bf16* Wc3   = (__bf16*)alloc((size_t)C_ * C_ * 2);
  __bf16* WN    = (__bf16*)alloc((size_t)N_ * C_ * 64 * 2);
  __bf16* ZROW  = (__bf16*)alloc((size_t)C_ * 2);        // all-zeros halo row
  float*  BLORA = (float*) alloc(64 * 4);
  float*  BSPE1 = (float*) alloc(32 * 4);
  float*  FS    = (float*) alloc((size_t)N_ * C_ * 4);
  float*  FSG   = (float*) alloc((size_t)N_ * C_ * 4);
  float*  FS2   = (float*) alloc((size_t)N_ * C4_ * 4);
  float*  MEANC = (float*) alloc((size_t)N_ * C4_ * 4);
  float*  ROUTE = (float*) alloc((size_t)N_ * E_ * 4);
  float*  PMIX  = (float*) alloc((size_t)N_ * 4);
  float*  BIASNC= (float*) alloc((size_t)N_ * C_ * 4);

  __bf16* Z     = (__bf16*)FEATC;   // [P,64]  (after featc mean consumed)
  __bf16* ZSPE  = (__bf16*)FEATG;   // [P,32]  (after featg consumed)
  __bf16* Ybf   = XT;               // [P,C]   (after conv1 consumed XT)
  float*  outp  = (float*)d_out;

  auto gemm = [&](const __bf16* A, int lda, const __bf16* Wm, int ldw, int wStride,
                  const float* bias, const float* biasNC,
                  float* Of, __bf16* Ob, int ldo, int Pr, int Co, int K, int nchw) {
    int colT = Co / 16;
    int ng = (colT % 4 == 0) ? 4 : ((colT % 2 == 0) ? 2 : 1);
    int tiles = (Pr / 16) * (colT / ng);
    int blocks = (tiles + 7) / 8;
    if (ng == 4)
      k_gemm<4><<<blocks, 256, 0, stream>>>(A, lda, Wm, ldw, wStride, bias, biasNC,
                                            Of, Ob, ldo, Pr, Co, K, nchw);
    else if (ng == 2)
      k_gemm<2><<<blocks, 256, 0, stream>>>(A, lda, Wm, ldw, wStride, bias, biasNC,
                                            Of, Ob, ldo, Pr, Co, K, nchw);
    else
      k_gemm<1><<<blocks, 256, 0, stream>>>(A, lda, Wm, ldw, wStride, bias, biasNC,
                                            Of, Ob, ldo, Pr, Co, K, nchw);
  };

  // --- stage inputs / weights as bf16 ---
  k_cvt_x<<<(N_ * C_ * HW_ + 255) / 256, 256, 0, stream>>>(x, XT);
  k_padcvt<<<(C_ * C_ + 255) / 256, 256, 0, stream>>>(conv1_w, C_, C_, Wc1, C_, C_);
  k_padcvt<<<(C4_ * C_ + 255) / 256, 256, 0, stream>>>(rc1_w, C4_, C_, Wrc1, C4_, C_);
  k_padcvt<<<(C4_ * C4_ + 255) / 256, 256, 0, stream>>>(rc3_w, C4_, C4_, Wrc3, C4_, C4_);
  k_padcvt<<<(64 * C_ + 255) / 256, 256, 0, stream>>>(lora_W, 48, C_, Wlora, 64, C_);
  k_padcvt<<<(32 * C_ + 255) / 256, 256, 0, stream>>>(spe_W, 24, C_, Wspe1, 32, C_);
  k_padcvt<<<(C_ * 32 + 255) / 256, 256, 0, stream>>>(spe_M, C_, 24, Wspe2, C_, 32);
  k_padcvt<<<(C_ * C_ + 255) / 256, 256, 0, stream>>>(conv3_w, C_, C_, Wc3, C_, C_);
  k_repack_shr<<<(9 * C_ * C_ + 255) / 256, 256, 0, stream>>>(share_w, Wshr);
  k_padf<<<1, 64, 0, stream>>>(lora_bW, 48, BLORA, 64);
  k_padf<<<1, 32, 0, stream>>>(spe_bW, 24, BSPE1, 32);
  k_padf<<<1, 192, 0, stream>>>(BLORA, 0, (float*)ZROW, 192);  // 768B of zeros

  // --- out_ori = conv1(x) ---
  gemm(XT, C_, Wc1, C_, 0, conv1_b, nullptr, nullptr, OObf, C_, P_, C_, C_, 0);

  // --- router feature branch ---
  gemm(OObf, C_, Wrc1, C_, 0, rc1_b, nullptr, BIGD, nullptr, C4_, P_, C4_, C_, 0);
  k_ln96_gelu<<<P_ / 8, 256, 0, stream>>>(BIGD, rln_g, rln_b, FEATG);
  gemm(FEATG, C4_, Wrc3, C4_, 0, rc3_b, nullptr, FEATC, nullptr, C4_, P_, C4_, C4_, 0);
  k_meanc<<<N_, C4_, 0, stream>>>(FEATC, MEANC);

  k_fs<<<N_, C_, 0, stream>>>(OObf, rsp_w, rsp_b, FS);
  k_fs_ln<<<N_, C_, 0, stream>>>(FS, rlnsp_g, rlnsp_b, FSG);
  k_fs2<<<N_, C4_, 0, stream>>>(FSG, rc2_w, rc2_b, FS2);

  k_router<<<N_, 32, 0, stream>>>(MEANC, FS2, router_w, router_b, ROUTE, PMIX, outp);

  // --- LoRA expert path ---
  gemm(OObf, C_, Wlora, C_, 0, BLORA, nullptr, nullptr, Z, 64, P_, 64, C_, 0);
  k_wn<<<(N_ * C_ * 64 + 255) / 256, 256, 0, stream>>>(lora_M, ROUTE, WN);
  k_biasnc<<<(N_ * C_ + 255) / 256, 256, 0, stream>>>(lora_bM, ROUTE, BIASNC);
  gemm(Z, 64, WN, 64, C_ * 64, nullptr, BIASNC, BIGD, nullptr, C_, P_, C_, 64, 0);

  // --- spe path ---
  gemm(OObf, C_, Wspe1, C_, 0, BSPE1, nullptr, nullptr, ZSPE, 32, P_, 32, C_, 0);
  gemm(ZSPE, 32, Wspe2, 32, 0, spe_bM, nullptr, SPE, nullptr, C_, P_, C_, 32, 0);

  // --- shared 3x3 conv (dominant op): 16x64 strip per wave, branchless halo ---
  {
    int tiles = (P_ / 16) * ((C_ / 16) / 4);
    k_conv3x3<<<(tiles + 7) / 8, 256, 0, stream>>>(OObf, Wshr, share_b, ZROW, SHR);
  }

  // --- combine + BN + GELU ---
  k_combine<<<(P_ * C_ + 255) / 256, 256, 0, stream>>>(
      BIGD, SPE, SHR, PMIX, bn_all_g, bn_all_b, bn_all_m, bn_all_v,
      bn_g, bn_b, bn_m, bn_v, Ybf);

  // --- final conv3, scatter to NCHW fp32 d_out ---
  gemm(Ybf, C_, Wc3, C_, 0, conv3_b, nullptr, outp, nullptr, C_, P_, C_, C_, 1);
}
